// GCNNet_54030688584325
// MI455X (gfx1250) — compile-verified, matrix operand order checked by reading, not run on hardware
//
#include <hip/hip_runtime.h>

// ---------------------------------------------------------------------------
// GCN (2x GCNConv + linear head) for MI455X / gfx1250.
// GEMMs: v_wmma_f32_16x16x32_bf16 (wave32), weights staged into LDS per
// workgroup via global_load_async_to_lds_b128 (ASYNCcnt) when available,
// read back with ds_load_b128. Scatter uses f32 atomics resident in 192MB L2.
// ---------------------------------------------------------------------------

typedef __bf16 bf16_t;
typedef __attribute__((ext_vector_type(16))) __bf16 v16bf;
typedef __attribute__((ext_vector_type(8)))  __bf16 v8bf;
typedef __attribute__((ext_vector_type(8)))  float  v8f;
typedef __attribute__((ext_vector_type(4)))  int    v4i;

#if defined(__gfx1250__) && __has_builtin(__builtin_amdgcn_global_load_async_to_lds_b128)
#define HAS_ASYNC_LDS 1
#else
#define HAS_ASYNC_LDS 0
#endif

__device__ __forceinline__ unsigned short f32_to_bf16_bits(float f) {
    unsigned u = __float_as_uint(f);
    u += 0x7FFFu + ((u >> 16) & 1u);     // round-to-nearest-even
    return (unsigned short)(u >> 16);
}

// ------------------------------ utility kernels ----------------------------

__global__ void k_zero_f32(float* __restrict__ p, long long n) {
    long long t = (long long)blockIdx.x * blockDim.x + threadIdx.x;
    if (t < n) p[t] = 0.0f;
}

__global__ void k_cvt_bf16(const float* __restrict__ in,
                           unsigned short* __restrict__ out, long long n) {
    long long t = (long long)blockIdx.x * blockDim.x + threadIdx.x;
    if (t < n) out[t] = f32_to_bf16_bits(in[t]);
}

// W is [K x Nc] row-major f32 -> Wt is [Nc x K] row-major bf16 (i.e. B^T)
__global__ void k_transpose_cvt(const float* __restrict__ W,
                                unsigned short* __restrict__ Wt,
                                int K, int Nc) {
    long long t = (long long)blockIdx.x * blockDim.x + threadIdx.x;
    if (t < (long long)K * Nc) {
        int k = (int)(t / Nc);
        int n = (int)(t % Nc);
        Wt[(long long)n * K + k] = f32_to_bf16_bits(W[t]);
    }
}

__global__ void k_edge_deg(const long long* __restrict__ dst,
                           float* __restrict__ deg, long long E) {
    long long t = (long long)blockIdx.x * blockDim.x + threadIdx.x;
    if (t < E) atomicAdd(&deg[dst[t]], 1.0f);
}

__global__ void k_degnorm(const float* __restrict__ deg,
                          float* __restrict__ dis,
                          float* __restrict__ dinv, long long n) {
    long long t = (long long)blockIdx.x * blockDim.x + threadIdx.x;
    if (t < n) {
        float d = deg[t] + 1.0f;          // self-loop in degree
        dis[t]  = rsqrtf(d);
        dinv[t] = 1.0f / d;
    }
}

// 32 lanes per edge; each lane moves a float4 of the 128 features.
__global__ void k_scatter(const float* __restrict__ h,
                          const long long* __restrict__ src,
                          const long long* __restrict__ dst,
                          const float* __restrict__ dis,
                          float* __restrict__ agg, long long E) {
    long long t = (long long)blockIdx.x * blockDim.x + threadIdx.x;
    if (t >= E * 32) return;
    long long e = t >> 5;
    int f = (int)(t & 31) * 4;
    long long s = src[e];
    long long d = dst[e];
    float norm = dis[s] * dis[d];
    const float4 hv = *(const float4*)(h + s * 128 + f);
    float* ap = agg + d * 128 + f;
    atomicAdd(ap + 0, hv.x * norm);
    atomicAdd(ap + 1, hv.y * norm);
    atomicAdd(ap + 2, hv.z * norm);
    atomicAdd(ap + 3, hv.w * norm);
}

// out = relu(agg + h * dinv + bias); emit bf16 (next GEMM) and optional f32.
__global__ void k_selfloop_relu(const float* __restrict__ agg,
                                const float* __restrict__ h,
                                const float* __restrict__ dinv,
                                const float* __restrict__ bias,
                                unsigned short* __restrict__ o16,
                                float* __restrict__ of32, long long total) {
    long long t = (long long)blockIdx.x * blockDim.x + threadIdx.x;
    if (t >= total) return;
    long long i = t >> 7;        // HID == 128
    int d = (int)(t & 127);
    float v = agg[t] + h[t] * dinv[i] + bias[d];
    v = fmaxf(v, 0.0f);
    o16[t] = f32_to_bf16_bits(v);
    if (of32) of32[t] = v;
}

// ------------------------------ WMMA GEMM ----------------------------------
// C[M x (16*NT)] = A_bf16[M x K] @ B_bf16[K x (16*NT)]  (Bt is B^T: row n = col n)
// 4 waves/block, one 16-row tile per wave, all NT column tiles per wave.
// B is staged chunk-by-chunk into LDS (async-to-LDS when available) and read
// with ds_load_b128; only A streams from global inside the MAC loop.
template <int K, int NT>
__global__ __launch_bounds__(128)
void k_wmma_gemm(const unsigned short* __restrict__ A_,
                 const unsigned short* __restrict__ Bt_,
                 const float* __restrict__ bias,
                 float* __restrict__ C, int mtiles, int ldc) {
    constexpr int KC = (K < 128) ? K : 128;        // K-chunk staged in LDS
    constexpr int ROWS = NT * 16;                  // B columns (rows of Bt)
    constexpr int UPR = KC / 8;                    // 16B units per row-chunk
    constexpr int UNITS = ROWS * UPR;              // 16B units per chunk
    __shared__ __align__(16) unsigned short Bs[ROWS * KC];

    const bf16_t* A = (const bf16_t*)A_;

    int wave = threadIdx.x >> 5;
    int lane = threadIdx.x & 31;
    int tile = blockIdx.x * 4 + wave;
    bool active = tile < mtiles;                   // uniform per wave
    int tclamp = active ? tile : (mtiles - 1);     // safe loads for tail waves

    int half = lane >> 4;                          // 0: K 0-7/16-23, 1: 8-15/24-31
    int l15  = lane & 15;
    int kb   = half * 8;

    const bf16_t* arow = A + (size_t)(tclamp * 16 + l15) * K + kb;

    v8f acc[NT];
#pragma unroll
    for (int i = 0; i < NT; i++)
#pragma unroll
        for (int j = 0; j < 8; j++) acc[i][j] = 0.0f;

    for (int kc = 0; kc < K; kc += KC) {
        // ---- cooperative stage of Bt[:, kc:kc+KC) into LDS ----
#pragma unroll
        for (int u = threadIdx.x; u < UNITS; u += 128) {
            int row = u / UPR;
            int cu  = u % UPR;
            const unsigned short* g = Bt_ + (size_t)row * K + kc + cu * 8;
            unsigned short* l = Bs + row * KC + cu * 8;
#if HAS_ASYNC_LDS
            __builtin_amdgcn_global_load_async_to_lds_b128(
                (__attribute__((address_space(1))) v4i*)(unsigned short*)g,
                (__attribute__((address_space(3))) v4i*)l, 0, 0);
#else
            *(uint4*)l = *(const uint4*)g;
#endif
        }
#if HAS_ASYNC_LDS
#if __has_builtin(__builtin_amdgcn_s_wait_asynccnt)
        __builtin_amdgcn_s_wait_asynccnt(0);
#else
        asm volatile("s_wait_asynccnt 0x0" ::: "memory");
#endif
#endif
        __syncthreads();

        // ---- MACs for this chunk ----
#pragma unroll
        for (int ks = 0; ks < KC; ks += 32) {
            const bf16_t* ap = arow + kc + ks;
            v8bf al = *(const v8bf*)(ap);
            v8bf ah = *(const v8bf*)(ap + 16);
            v16bf a;
#pragma unroll
            for (int i = 0; i < 8; i++) { a[i] = al[i]; a[i + 8] = ah[i]; }

#pragma unroll
            for (int nt = 0; nt < NT; nt++) {
                const unsigned short* bp = Bs + (nt * 16 + l15) * KC + ks + kb;
                v8bf bl = *(const v8bf*)(bp);
                v8bf bh = *(const v8bf*)(bp + 16);
                v16bf b;
#pragma unroll
                for (int i = 0; i < 8; i++) { b[i] = bl[i]; b[i + 8] = bh[i]; }
                acc[nt] = __builtin_amdgcn_wmma_f32_16x16x32_bf16(
                    false, a, false, b, (short)0, acc[nt], false, false);
            }
        }
        __syncthreads();   // LDS reused by next chunk
    }

    if (active) {
        // C/D layout: lanes 0-15 -> M=r, lanes 16-31 -> M=r+8, N = lane&15
        int mrow = tile * 16 + half * 8;
#pragma unroll
        for (int nt = 0; nt < NT; nt++) {
            int n = nt * 16 + l15;
            float badd = bias ? bias[n] : 0.0f;
#pragma unroll
            for (int r = 0; r < 8; r++) {
                C[(size_t)(mrow + r) * ldc + n] = acc[nt][r] + badd;
            }
        }
    }
}

// ------------------------------ launcher -----------------------------------

extern "C" void kernel_launch(void* const* d_in, const int* in_sizes, int n_in,
                              void* d_out, int out_size, void* d_ws, size_t ws_size,
                              hipStream_t stream) {
    const float*     x  = (const float*)d_in[0];
    const long long* ei = (const long long*)d_in[1];   // int64 edge_index [2, E]
    const float*     W1 = (const float*)d_in[2];
    const float*     b1 = (const float*)d_in[3];
    const float*     W2 = (const float*)d_in[4];
    const float*     b2 = (const float*)d_in[5];
    const float*     Wl = (const float*)d_in[6];
    const float*     bl = (const float*)d_in[7];

    const int IN_DIM = 512, HID = 128, OUT = 16;
    const long long N = in_sizes[0] / IN_DIM;
    const long long E = in_sizes[1] / 2;
    const long long* src = ei;
    const long long* dst = ei + E;

    // workspace carve-out (256B aligned)
    char* ws = (char*)d_ws;
    size_t off = 0;
    auto carve = [&](size_t bytes) -> char* {
        char* p = ws + off;
        off = (off + bytes + 255) & ~(size_t)255;
        return p;
    };
    float*          deg  = (float*)carve((size_t)N * 4);
    float*          dis  = (float*)carve((size_t)N * 4);
    float*          dinv = (float*)carve((size_t)N * 4);
    unsigned short* x16  = (unsigned short*)carve((size_t)N * IN_DIM * 2);
    unsigned short* w1t  = (unsigned short*)carve((size_t)IN_DIM * HID * 2);
    unsigned short* w2t  = (unsigned short*)carve((size_t)HID * HID * 2);
    unsigned short* wlt  = (unsigned short*)carve((size_t)HID * OUT * 2);
    float*          h    = (float*)carve((size_t)N * HID * 4);   // h1, then h2
    float*          agg  = (float*)carve((size_t)N * HID * 4);   // agg1, then agg2
    unsigned short* r16  = (unsigned short*)carve((size_t)N * HID * 2);

    float* out_head = (float*)d_out;                    // [N, 16]
    float* out_emb  = (float*)d_out + (size_t)N * OUT;  // [N, 128]

    const int TB = 256;
    auto blocks = [&](long long n) { return (unsigned)((n + TB - 1) / TB); };

    // degrees + normalization factors
    k_zero_f32<<<blocks(N), TB, 0, stream>>>(deg, N);
    k_edge_deg<<<blocks(E), TB, 0, stream>>>(dst, deg, E);
    k_degnorm<<<blocks(N), TB, 0, stream>>>(deg, dis, dinv, N);

    // bf16 staging (x read once anyway; weights tiny, transposed for WMMA B)
    k_cvt_bf16<<<blocks(N * IN_DIM), TB, 0, stream>>>(x, x16, N * IN_DIM);
    k_transpose_cvt<<<blocks((long long)IN_DIM * HID), TB, 0, stream>>>(W1, w1t, IN_DIM, HID);
    k_transpose_cvt<<<blocks((long long)HID * HID), TB, 0, stream>>>(W2, w2t, HID, HID);
    k_transpose_cvt<<<blocks((long long)HID * OUT), TB, 0, stream>>>(Wl, wlt, HID, OUT);

    int mt = (int)(N / 16);              // N = 100000 -> 6250 tiles
    unsigned gb = (unsigned)((mt + 3) / 4);

    // ----- layer 1: h = x @ W1 ; agg = scatter + self-loop ; relu -----
    k_wmma_gemm<512, 8><<<gb, 128, 0, stream>>>(x16, w1t, nullptr, h, mt, HID);
    k_zero_f32<<<blocks(N * HID), TB, 0, stream>>>(agg, N * HID);
    k_scatter<<<blocks(E * 32), TB, 0, stream>>>(h, src, dst, dis, agg, E);
    k_selfloop_relu<<<blocks(N * HID), TB, 0, stream>>>(agg, h, dinv, b1, r16,
                                                        nullptr, N * HID);

    // ----- layer 2 -----
    k_wmma_gemm<128, 8><<<gb, 128, 0, stream>>>(r16, w2t, nullptr, h, mt, HID);
    k_zero_f32<<<blocks(N * HID), TB, 0, stream>>>(agg, N * HID);
    k_scatter<<<blocks(E * 32), TB, 0, stream>>>(h, src, dst, dis, agg, E);
    k_selfloop_relu<<<blocks(N * HID), TB, 0, stream>>>(agg, h, dinv, b2, r16,
                                                        out_emb, N * HID);

    // ----- head: out = emb @ Wl + bl -----
    k_wmma_gemm<128, 1><<<gb, 128, 0, stream>>>(r16, wlt, bl, out_head, mt, OUT);
}
